// Decoder_5781025980829
// MI455X (gfx1250) — matrix operand order repaired and checked by reading, not compiled, not run on hardware
//
#include <hip/hip_runtime.h>
#include <hip/hip_bf16.h>

#define B_   32
#define T_   64
#define V_   32000
#define E_   256
#define H_   512
#define U_   768          // H + E
#define G3_  2304         // 3 * U
#define BT_  2048         // B * T

typedef _Float16 h8  __attribute__((ext_vector_type(8)));
typedef _Float16 h16 __attribute__((ext_vector_type(16)));
typedef float    f8  __attribute__((ext_vector_type(8)));

union ABfrag { h16 v; h8 h[2]; };

// ---------------------------------------------------------------------------
// Fragment layout (CDNA5 16-bit WMMA, wave32):
//   A 16x32: lane<16 -> row=lane, K {0..7,16..23}; lane>=16 -> K {8..15,24..31}
//   B 32x16 via BT (NxK): lane<16 -> col=lane, K {0..7,16..23}; lane>=16 -> col=lane-16, K {8..15,24..31}
// With BT stored row-major (N x K) every fragment load is a contiguous 16B.
// ---------------------------------------------------------------------------

// A is fp32 in memory (live GRU state), converted on the fly.
__device__ __forceinline__ void wmma_k_f32a(const float* __restrict__ arow,
                                            const _Float16* __restrict__ brow,
                                            int K, f8& acc) {
  for (int k0 = 0; k0 < K; k0 += 32) {
    ABfrag a, b;
#pragma unroll
    for (int j = 0; j < 8; ++j) {
      a.v[j]     = (_Float16)arow[k0 + j];
      a.v[j + 8] = (_Float16)arow[k0 + 16 + j];
    }
    b.h[0] = *(const h8*)(brow + k0);
    b.h[1] = *(const h8*)(brow + k0 + 16);
    acc = __builtin_amdgcn_wmma_f32_16x16x32_f16(false, a.v, false, b.v,
                                                 (short)0, acc, false, false);
  }
}

// ---------------------------------------------------------------------------
// Weight prep: fp32 (R x C) -> transposed f16 (C x R)
// ---------------------------------------------------------------------------
__global__ void k_transpose_to_f16(const float* __restrict__ in,
                                   _Float16* __restrict__ out, int R, int C) {
  long i = (long)blockIdx.x * blockDim.x + threadIdx.x;
  if (i >= (long)R * C) return;
  int r = (int)(i / C), c = (int)(i % C);
  out[(long)c * R + r] = (_Float16)in[i];
}

// state0 = initial_state @ fc_w + fc_b   (32x512 @ 512x768) -- tiny, scalar
__global__ void k_state0(const float* __restrict__ st, const float* __restrict__ fcw,
                         const float* __restrict__ fcb, float* __restrict__ h0) {
  int i = blockIdx.x * blockDim.x + threadIdx.x;
  if (i >= B_ * U_) return;
  int b = i / U_, u = i % U_;
  float acc = fcb[u];
  for (int j = 0; j < H_; ++j) acc += st[b * H_ + j] * fcw[j * U_ + u];
  h0[i] = acc;
}

// Attention is timestep-invariant (softmax shift invariance):
// alpha = softmax(enc @ w_e + b), ctx[b,h] = sum_t alpha[b,t]*enc[b,t,h]
__global__ void k_attention(const float* __restrict__ enc,
                            const float* __restrict__ attw,
                            const float* __restrict__ attb,
                            float* __restrict__ ctx) {
  __shared__ float red[256];
  __shared__ float alpha[T_];
  int b = blockIdx.x, tid = threadIdx.x;
  int t = tid & 63, part = tid >> 6;              // 4 partials of 128
  const float* e = enc + ((long)b * T_ + t) * H_;
  float s = 0.f;
  for (int j = part * 128; j < part * 128 + 128; ++j) s += e[j] * attw[j];
  red[tid] = s;
  __syncthreads();
  if (tid < T_)
    alpha[tid] = red[tid] + red[tid + 64] + red[tid + 128] + red[tid + 192] + attb[0];
  __syncthreads();
  if (tid == 0) {
    float m = alpha[0];
    for (int k = 1; k < T_; ++k) m = fmaxf(m, alpha[k]);
    float sum = 0.f;
    for (int k = 0; k < T_; ++k) { alpha[k] = expf(alpha[k] - m); sum += alpha[k]; }
    float inv = 1.f / sum;
    for (int k = 0; k < T_; ++k) alpha[k] *= inv;
  }
  __syncthreads();
  for (int hh = tid; hh < H_; hh += 256) {
    float acc = 0.f;
    for (int k = 0; k < T_; ++k)
      acc += alpha[k] * enc[((long)b * T_ + k) * H_ + hh];
    ctx[b * H_ + hh] = acc;
  }
}

// X[b*T+t] = [embedding[tok[b,t]] , ctx[b]]  as f16  (2048 x 768)
__global__ void k_build_x(const int* __restrict__ toks, const float* __restrict__ emb,
                          const float* __restrict__ ctx, _Float16* __restrict__ X) {
  long i = (long)blockIdx.x * blockDim.x + threadIdx.x;
  if (i >= (long)BT_ * U_) return;
  int r = (int)(i / U_), c = (int)(i % U_);
  int b = r / T_, t = r % T_;
  float v = (c < E_) ? emb[(long)toks[b * T_ + t] * E_ + c]
                     : ctx[b * H_ + (c - E_)];
  X[i] = (_Float16)v;
}

// ---------------------------------------------------------------------------
// 4x4 register-blocked WMMA GEMM: C(MxN,f32) = A(MxK,f16)@BT(NxK,f16)^T + bias
// One wave owns a 64x64 macro-tile: per K-step, 8 fragment loads feed 16
// WMMAs (4x operand reuse in registers vs one-tile-per-wave).
// Grid: exactly (tilesM/4)*(tilesN/4) waves, 4 waves (128 thr) per block.
// ---------------------------------------------------------------------------
__global__ void __launch_bounds__(128)
k_gemm_f16_blk(const _Float16* __restrict__ A, const _Float16* __restrict__ BT,
               const float* __restrict__ bias, float* __restrict__ C,
               int nBlkN, int N, int K) {
  int wave = threadIdx.x >> 5, lane = threadIdx.x & 31;
  int blk = blockIdx.x * 4 + wave;
  int bm = blk / nBlkN, bn = blk % nBlkN;
  int half_id = lane >> 4, l16 = lane & 15;

  const _Float16* arow[4];
  const _Float16* brow[4];
#pragma unroll
  for (int i = 0; i < 4; ++i) {
    arow[i] = A + (long)(bm * 64 + i * 16 + l16) * K + 8 * half_id;
    brow[i] = BT + (long)(bn * 64 + i * 16 + l16) * K + 8 * half_id;
  }

  f8 acc[16];
#pragma unroll
  for (int j = 0; j < 4; ++j) {
    float bv = bias ? bias[bn * 64 + j * 16 + l16] : 0.f;
#pragma unroll
    for (int i = 0; i < 4; ++i)
#pragma unroll
      for (int e = 0; e < 8; ++e) acc[i * 4 + j][e] = bv;
  }

  for (int k0 = 0; k0 < K; k0 += 32) {
    ABfrag a[4], b[4];
#pragma unroll
    for (int i = 0; i < 4; ++i) {
      a[i].h[0] = *(const h8*)(arow[i] + k0);
      a[i].h[1] = *(const h8*)(arow[i] + k0 + 16);
      b[i].h[0] = *(const h8*)(brow[i] + k0);
      b[i].h[1] = *(const h8*)(brow[i] + k0 + 16);
    }
#pragma unroll
    for (int i = 0; i < 4; ++i)
#pragma unroll
      for (int j = 0; j < 4; ++j)
        acc[i * 4 + j] = __builtin_amdgcn_wmma_f32_16x16x32_f16(
            false, a[i].v, false, b[j].v, (short)0, acc[i * 4 + j], false, false);
  }

#pragma unroll
  for (int i = 0; i < 4; ++i)
#pragma unroll
    for (int j = 0; j < 4; ++j) {
      int col = bn * 64 + j * 16 + l16;
#pragma unroll
      for (int e = 0; e < 8; ++e)
        C[(long)(bm * 64 + i * 16 + e + 8 * half_id) * N + col] = acc[i * 4 + j][e];
    }
}

// One GRU step: gh = h@gru_wr (WMMA), gates, h_new.  48 blocks x 192 thr.
// Block owns u-slice [u0,u0+16) for all 32 batch rows and all 3 gates:
// wave w -> gate g=w/2, M-tile m=w%2.  gh tiles staged in LDS, then gate math.
__global__ void __launch_bounds__(192)
k_gru_step(int t, const float* __restrict__ hcur, float* __restrict__ hnext,
           const _Float16* __restrict__ WrT, const float* __restrict__ GX,
           const float* __restrict__ br, _Float16* __restrict__ Hf16) {
  __shared__ float ghl[3][32][16];
  int u0 = blockIdx.x * 16;
  int tid = threadIdx.x, wave = tid >> 5, lane = tid & 31;
  int g = wave >> 1, m = wave & 1;
  int half_id = lane >> 4, l16 = lane & 15;

  const float*    arow = hcur + (m * 16 + l16) * U_ + 8 * half_id;
  const _Float16* brow = WrT + (long)(g * U_ + u0 + l16) * U_ + 8 * half_id;
  f8 acc;
#pragma unroll
  for (int i = 0; i < 8; ++i) acc[i] = 0.f;
  wmma_k_f32a(arow, brow, U_, acc);
#pragma unroll
  for (int i = 0; i < 8; ++i)
    ghl[g][m * 16 + i + 8 * half_id][l16] = acc[i];
  __syncthreads();

  for (int e = tid; e < 32 * 16; e += 192) {
    int b = e >> 4, ul = e & 15, u = u0 + ul;
    long r = (long)b * T_ + t;
    float gxz = GX[r * G3_ + u];
    float gxr = GX[r * G3_ + U_ + u];
    float gxh = GX[r * G3_ + 2 * U_ + u];
    float z  = 1.f / (1.f + expf(-(gxz + ghl[0][b][ul] + br[u])));
    float rg = 1.f / (1.f + expf(-(gxr + ghl[1][b][ul] + br[U_ + u])));
    float hc = tanhf(gxh + rg * (ghl[2][b][ul] + br[2 * U_ + u]));
    float ho = hcur[b * U_ + u];
    float hn = z * ho + (1.f - z) * hc;
    hnext[b * U_ + u] = hn;
    Hf16[r * U_ + u] = (_Float16)hn;
  }
}

// In-place row softmax over V=32000, one block per row (logits live in d_out).
__global__ void __launch_bounds__(256)
k_softmax_rows(float* __restrict__ out) {
  __shared__ float red[256];
  float* p = out + (long)blockIdx.x * V_;
  int tid = threadIdx.x;
  float m = -1e30f;
  for (int v = tid; v < V_; v += 256) m = fmaxf(m, p[v]);
  red[tid] = m; __syncthreads();
  for (int s = 128; s > 0; s >>= 1) {
    if (tid < s) red[tid] = fmaxf(red[tid], red[tid + s]);
    __syncthreads();
  }
  m = red[0]; __syncthreads();
  float sum = 0.f;
  for (int v = tid; v < V_; v += 256) sum += expf(p[v] - m);
  red[tid] = sum; __syncthreads();
  for (int s = 128; s > 0; s >>= 1) {
    if (tid < s) red[tid] += red[tid + s];
    __syncthreads();
  }
  float inv = 1.f / red[0];
  for (int v = tid; v < V_; v += 256) p[v] = expf(p[v] - m) * inv;
}

// ---------------------------------------------------------------------------
extern "C" void kernel_launch(void* const* d_in, const int* in_sizes, int n_in,
                              void* d_out, int out_size, void* d_ws, size_t ws_size,
                              hipStream_t stream) {
  const int*   toks    = (const int*)d_in[0];
  const float* istate  = (const float*)d_in[1];
  const float* enc     = (const float*)d_in[2];
  const float* emb     = (const float*)d_in[3];
  const float* fc_w    = (const float*)d_in[4];
  const float* fc_b    = (const float*)d_in[5];
  const float* att_w   = (const float*)d_in[6];
  const float* att_b   = (const float*)d_in[7];
  const float* gru_wk  = (const float*)d_in[8];
  const float* gru_wr  = (const float*)d_in[9];
  const float* gru_bi  = (const float*)d_in[10];
  const float* gru_br  = (const float*)d_in[11];
  const float* dense_w = (const float*)d_in[12];
  const float* dense_b = (const float*)d_in[13];
  float* out = (float*)d_out;

  char* ws = (char*)d_ws;
  size_t off = 0;
  auto alloc = [&](size_t bytes) -> char* {
    char* p = ws + off;
    off += (bytes + 255) & ~(size_t)255;
    return p;
  };
  _Float16* WkT   = (_Float16*)alloc((size_t)G3_ * U_ * 2);   // 3.5 MB
  _Float16* WrT   = (_Float16*)alloc((size_t)G3_ * U_ * 2);   // 3.5 MB
  _Float16* DwT   = (_Float16*)alloc((size_t)V_ * U_ * 2);    // 49.2 MB
  _Float16* X     = (_Float16*)alloc((size_t)BT_ * U_ * 2);   // 3.1 MB
  _Float16* Hf16  = (_Float16*)alloc((size_t)BT_ * U_ * 2);   // 3.1 MB
  float*    GX    = (float*)alloc((size_t)BT_ * G3_ * 4);     // 18.9 MB
  float*    ctx   = (float*)alloc((size_t)B_ * H_ * 4);
  float*    hbuf0 = (float*)alloc((size_t)B_ * U_ * 4);
  float*    hbuf1 = (float*)alloc((size_t)B_ * U_ * 4);

  // 1) f16 transposed weights (BT layout => contiguous WMMA fragments)
  k_transpose_to_f16<<<(G3_ * U_ + 255) / 256, 256, 0, stream>>>(gru_wk, WkT, U_, G3_);
  k_transpose_to_f16<<<(G3_ * U_ + 255) / 256, 256, 0, stream>>>(gru_wr, WrT, U_, G3_);
  k_transpose_to_f16<<<(int)(((long)V_ * U_ + 255) / 256), 256, 0, stream>>>(dense_w, DwT, U_, V_);

  // 2) precompute: state0, time-invariant attention context, X, GX
  k_state0<<<(B_ * U_ + 255) / 256, 256, 0, stream>>>(istate, fc_w, fc_b, hbuf0);
  k_attention<<<B_, 256, 0, stream>>>(enc, att_w, att_b, ctx);
  k_build_x<<<(int)(((long)BT_ * U_ + 255) / 256), 256, 0, stream>>>(toks, emb, ctx, X);
  // GX = X @ gru_wk + bi : macro-tiles (128/4) x (144/4) = 32 x 36
  k_gemm_f16_blk<<<(32 * 36) / 4, 128, 0, stream>>>(X, WkT, gru_bi, GX, 36, G3_, U_);

  // 3) sequential GRU recurrence (ping-pong h across launches)
  for (int t = 0; t < T_; ++t) {
    const float* hc = (t & 1) ? hbuf1 : hbuf0;
    float*       hn = (t & 1) ? hbuf0 : hbuf1;
    k_gru_step<<<48, 192, 0, stream>>>(t, hc, hn, WrT, GX, gru_br, Hf16);
  }

  // 4) logits = H @ dense_w + b : macro-tiles (128/4) x (2000/4) = 32 x 500
  k_gemm_f16_blk<<<(32 * 500) / 4, 128, 0, stream>>>(Hf16, DwT, dense_b, out, 500, V_, U_);

  // 5) row softmax in place
  k_softmax_rows<<<BT_, 256, 0, stream>>>(out);
}
